// OTLoss_1709396984027
// MI455X (gfx1250) — compile-verified, matrix operand order checked by reading, not run on hardware
//
#include <hip/hip_runtime.h>
#include <hip/hip_bf16.h>
#include <math.h>

// Problem constants (from reference)
#define Bn 16
#define Sn 1024
#define Dn 2048
#define EPSF      0.1f
#define INV_EPS   10.0f
#define NEG_LOG_N (-6.93147180559945f)   // -log(1024)
#define NCHUNK    16                     // row chunks in g-pass (64 rows each)

typedef __attribute__((ext_vector_type(16))) _Float16 v16h;
typedef __attribute__((ext_vector_type(8)))  _Float16 v8h;
typedef __attribute__((ext_vector_type(8)))  float    v8f;

// ---------------------------------------------------------------------------
// helpers
// ---------------------------------------------------------------------------
__device__ __forceinline__ void lse_add(float& m, float& s, float x) {
    if (x > m) { s = s * __expf(m - x) + 1.0f; m = x; }
    else       { s += __expf(x - m); }
}
__device__ __forceinline__ void lse_merge(float& m, float& s, float m2, float s2) {
    if (m2 > m) { s = s * __expf(m - m2) + s2; m = m2; }
    else        { s = s + s2 * __expf(m2 - m); }
}

// Load one 16-bit A/B WMMA fragment (16x32 / 32x16) per the CDNA5 layout:
// lane<16: K = kk..kk+7 (v0-3) and kk+16..kk+23 (v4-7); lane>=16: kk = +8.
__device__ __forceinline__ v16h ldfrag(const _Float16* p) {
    union { v16h v; v8h h[2]; } u;
    u.h[0] = *(const v8h*)(p);
    u.h[1] = *(const v8h*)(p + 16);
    return u.v;
}

// ---------------------------------------------------------------------------
// 1) Row normalize info: f16 copy + reciprocal norm per row. 1 wave per row.
// ---------------------------------------------------------------------------
__global__ __launch_bounds__(256) void norm_convert(
    const float* __restrict__ in, _Float16* __restrict__ out,
    float* __restrict__ rn)
{
    const int wid = threadIdx.x >> 5, lane = threadIdx.x & 31;
    const int row = blockIdx.x * 8 + wid;               // [0, Bn*Sn)
    const float4* p = (const float4*)(in + (size_t)row * Dn);
    _Float16*     q = out + (size_t)row * Dn;
    float ss = 0.f;
    for (int t = lane; t < Dn / 4; t += 32) {
        float4 x = p[t];
        const int d = t * 4;
        q[d + 0] = (_Float16)x.x;
        q[d + 1] = (_Float16)x.y;
        q[d + 2] = (_Float16)x.z;
        q[d + 3] = (_Float16)x.w;
        ss += x.x * x.x + x.y * x.y + x.z * x.z + x.w * x.w;
    }
    #pragma unroll
    for (int off = 16; off; off >>= 1) ss += __shfl_xor(ss, off, 32);
    if (lane == 0) rn[row] = 1.0f / fmaxf(__builtin_sqrtf(ss), 1e-12f);
}

// ---------------------------------------------------------------------------
// 2) Batched cost GEMM: C[b,i,j] = 0.5 - 0.5 * (t_i . s_j) * rnT_i * rnS_j
//    Wave computes 32x32 output tile via 2x2 v_wmma_f32_16x16x32_f16.
//    Grid: (Sn/128, Sn/64, Bn), block 256 (8 waves: 4 in j, 2 in i).
// ---------------------------------------------------------------------------
__global__ __launch_bounds__(256) void cost_gemm(
    const _Float16* __restrict__ T16, const _Float16* __restrict__ S16,
    const float* __restrict__ rnT, const float* __restrict__ rnS,
    float* __restrict__ C)
{
    const int b    = blockIdx.z;
    const int lane = threadIdx.x & 31;
    const int wid  = threadIdx.x >> 5;
    const int wx   = wid & 3;            // 0..3  -> j
    const int wy   = wid >> 2;           // 0..1  -> i
    const int i0   = blockIdx.y * 64  + wy * 32;
    const int j0   = blockIdx.x * 128 + wx * 32;

    const _Float16* Tb = T16 + (size_t)b * Sn * Dn;
    const _Float16* Sb = S16 + (size_t)b * Sn * Dn;

    const int l15    = lane & 15;
    const int kk_off = (lane >> 4) << 3;  // 0 or 8

    const _Float16* ta0 = Tb + (size_t)(i0 +      l15) * Dn + kk_off;
    const _Float16* ta1 = Tb + (size_t)(i0 + 16 + l15) * Dn + kk_off;
    const _Float16* sb0 = Sb + (size_t)(j0 +      l15) * Dn + kk_off;
    const _Float16* sb1 = Sb + (size_t)(j0 + 16 + l15) * Dn + kk_off;

    v8f acc00 = {}, acc01 = {}, acc10 = {}, acc11 = {};

    #pragma unroll 2
    for (int k0 = 0; k0 < Dn; k0 += 32) {
        v16h a0 = ldfrag(ta0 + k0);
        v16h a1 = ldfrag(ta1 + k0);
        v16h b0 = ldfrag(sb0 + k0);
        v16h b1 = ldfrag(sb1 + k0);
        acc00 = __builtin_amdgcn_wmma_f32_16x16x32_f16(false, a0, false, b0,
                                                       (short)0, acc00, false, false);
        acc01 = __builtin_amdgcn_wmma_f32_16x16x32_f16(false, a0, false, b1,
                                                       (short)0, acc01, false, false);
        acc10 = __builtin_amdgcn_wmma_f32_16x16x32_f16(false, a1, false, b0,
                                                       (short)0, acc10, false, false);
        acc11 = __builtin_amdgcn_wmma_f32_16x16x32_f16(false, a1, false, b1,
                                                       (short)0, acc11, false, false);
    }

    // Epilogue: C/D layout — VGPR r, lane l: row = r + 8*(l>>4), col = l&15.
    const int hi = (lane >> 4) << 3;     // 0 or 8
    float* Cb = C + ((size_t)b * Sn) * Sn;
    v8f* accs[2][2] = { { &acc00, &acc01 }, { &acc10, &acc11 } };
    #pragma unroll
    for (int ta = 0; ta < 2; ++ta) {
        #pragma unroll
        for (int tb = 0; tb < 2; ++tb) {
            const int gj = j0 + tb * 16 + l15;
            const float rj = rnS[b * Sn + gj];
            v8f a = *accs[ta][tb];
            #pragma unroll
            for (int r = 0; r < 8; ++r) {
                const int gi = i0 + ta * 16 + hi + r;
                const float ri = rnT[b * Sn + gi];
                Cb[(size_t)gi * Sn + gj] = 0.5f - 0.5f * a[r] * ri * rj;
            }
        }
    }
}

// ---------------------------------------------------------------------------
// 3) init f = g = 0   (f and g contiguous: 2*Bn*Sn floats)
// ---------------------------------------------------------------------------
__global__ __launch_bounds__(256) void init_fg(float* __restrict__ fg) {
    int i = blockIdx.x * 256 + threadIdx.x;
    if (i < 2 * Bn * Sn) fg[i] = 0.0f;
}

// ---------------------------------------------------------------------------
// 4a) f-update: f_i = eps*(loga - LSE_j((g_j - C_ij)/eps)). 1 wave per row,
//     float4 (b128) streaming of the row.
// ---------------------------------------------------------------------------
__global__ __launch_bounds__(256) void f_update(
    const float* __restrict__ C, const float* __restrict__ g,
    float* __restrict__ f)
{
    const int wid = threadIdx.x >> 5, lane = threadIdx.x & 31;
    const int row = blockIdx.x * 8 + wid;         // b*Sn + i
    const int b   = row >> 10;
    const float4* Cr4 = (const float4*)(C + ((size_t)row << 10));
    const float4* g4  = (const float4*)(g + (b << 10));
    float m = -INFINITY, s = 0.0f;
    for (int t = lane; t < Sn / 4; t += 32) {
        float4 c  = Cr4[t];
        float4 gg = g4[t];
        lse_add(m, s, (gg.x - c.x) * INV_EPS);
        lse_add(m, s, (gg.y - c.y) * INV_EPS);
        lse_add(m, s, (gg.z - c.z) * INV_EPS);
        lse_add(m, s, (gg.w - c.w) * INV_EPS);
    }
    #pragma unroll
    for (int off = 16; off; off >>= 1)
        lse_merge(m, s, __shfl_xor(m, off, 32), __shfl_xor(s, off, 32));
    if (lane == 0) f[row] = EPSF * (NEG_LOG_N - (m + __logf(s)));
}

// ---------------------------------------------------------------------------
// 4b) g partial: per (batch, 64-row chunk), each thread owns 4 consecutive
//     columns and streams one float4 per row (coalesced b128).
//     Grid: Bn*NCHUNK = 256 blocks of 256 threads.
// ---------------------------------------------------------------------------
__global__ __launch_bounds__(256) void g_partial(
    const float* __restrict__ C, const float* __restrict__ f,
    float* __restrict__ gpm, float* __restrict__ gps)
{
    const int chunk = blockIdx.x & (NCHUNK - 1);
    const int b     = blockIdx.x / NCHUNK;
    const int t     = threadIdx.x;                 // float4 column index
    const float*  fb  = f + (b << 10);
    const float4* Cb4 = (const float4*)(C + ((size_t)b << 20));
    float m0 = -INFINITY, m1 = -INFINITY, m2 = -INFINITY, m3 = -INFINITY;
    float s0 = 0.f, s1 = 0.f, s2 = 0.f, s3 = 0.f;
    const int i0 = chunk * (Sn / NCHUNK);
    for (int i = i0; i < i0 + Sn / NCHUNK; ++i) {
        float4 c = Cb4[((size_t)i << 8) + t];      // 256 float4 per row
        const float fi = fb[i];
        lse_add(m0, s0, (fi - c.x) * INV_EPS);
        lse_add(m1, s1, (fi - c.y) * INV_EPS);
        lse_add(m2, s2, (fi - c.z) * INV_EPS);
        lse_add(m3, s3, (fi - c.w) * INV_EPS);
    }
    const int o = ((b * NCHUNK + chunk) << 8) + t; // float4 index
    ((float4*)gpm)[o] = make_float4(m0, m1, m2, m3);
    ((float4*)gps)[o] = make_float4(s0, s1, s2, s3);
}

// 4c) g combine: merge NCHUNK partials -> g_j (4 cols/thread, float4 I/O).
//     Grid: Bn blocks of 256 threads.
__global__ __launch_bounds__(256) void g_combine(
    const float* __restrict__ gpm, const float* __restrict__ gps,
    float* __restrict__ g)
{
    const int b = blockIdx.x;
    const int t = threadIdx.x;
    float m0 = -INFINITY, m1 = -INFINITY, m2 = -INFINITY, m3 = -INFINITY;
    float s0 = 0.f, s1 = 0.f, s2 = 0.f, s3 = 0.f;
    #pragma unroll
    for (int c = 0; c < NCHUNK; ++c) {
        const int o = ((b * NCHUNK + c) << 8) + t;
        float4 mm = ((const float4*)gpm)[o];
        float4 ss = ((const float4*)gps)[o];
        lse_merge(m0, s0, mm.x, ss.x);
        lse_merge(m1, s1, mm.y, ss.y);
        lse_merge(m2, s2, mm.z, ss.z);
        lse_merge(m3, s3, mm.w, ss.w);
    }
    float4 go;
    go.x = EPSF * (NEG_LOG_N - (m0 + __logf(s0)));
    go.y = EPSF * (NEG_LOG_N - (m1 + __logf(s1)));
    go.z = EPSF * (NEG_LOG_N - (m2 + __logf(s2)));
    go.w = EPSF * (NEG_LOG_N - (m3 + __logf(s3)));
    ((float4*)g)[(b << 8) + t] = go;
}

// ---------------------------------------------------------------------------
// 5) loss per row:  rowpart[row] = sum_j exp((f_i+g_j-C_ij)/eps) * C_ij
// ---------------------------------------------------------------------------
__global__ __launch_bounds__(256) void loss_rows(
    const float* __restrict__ C, const float* __restrict__ f,
    const float* __restrict__ g, float* __restrict__ rowpart)
{
    const int wid = threadIdx.x >> 5, lane = threadIdx.x & 31;
    const int row = blockIdx.x * 8 + wid;
    const int b   = row >> 10;
    const float4* Cr4 = (const float4*)(C + ((size_t)row << 10));
    const float4* g4  = (const float4*)(g + (b << 10));
    const float fi = f[row];
    float acc = 0.0f;
    for (int t = lane; t < Sn / 4; t += 32) {
        float4 c  = Cr4[t];
        float4 gg = g4[t];
        acc += __expf((fi + gg.x - c.x) * INV_EPS) * c.x;
        acc += __expf((fi + gg.y - c.y) * INV_EPS) * c.y;
        acc += __expf((fi + gg.z - c.z) * INV_EPS) * c.z;
        acc += __expf((fi + gg.w - c.w) * INV_EPS) * c.w;
    }
    #pragma unroll
    for (int off = 16; off; off >>= 1) acc += __shfl_xor(acc, off, 32);
    if (lane == 0) rowpart[row] = acc;
}

// 6) final: out = (sum of rowparts) / Bn
__global__ __launch_bounds__(256) void loss_final(
    const float* __restrict__ rowpart, float* __restrict__ out)
{
    __shared__ float sm[256];
    float acc = 0.0f;
    for (int i = threadIdx.x; i < Bn * Sn; i += 256) acc += rowpart[i];
    sm[threadIdx.x] = acc;
    __syncthreads();
    for (int off = 128; off; off >>= 1) {
        if (threadIdx.x < off) sm[threadIdx.x] += sm[threadIdx.x + off];
        __syncthreads();
    }
    if (threadIdx.x == 0) out[0] = sm[0] / (float)Bn;
}

// ---------------------------------------------------------------------------
// launcher
// ---------------------------------------------------------------------------
extern "C" void kernel_launch(void* const* d_in, const int* in_sizes, int n_in,
                              void* d_out, int out_size, void* d_ws, size_t ws_size,
                              hipStream_t stream) {
    (void)in_sizes; (void)n_in; (void)out_size; (void)ws_size;
    const float* student = (const float*)d_in[0];   // [B,S,D] f32
    const float* teacher = (const float*)d_in[1];   // [B,S,D] f32
    float* out = (float*)d_out;                     // scalar f32

    // workspace layout (all offsets 64B-aligned by construction)
    char* w = (char*)d_ws;
    _Float16* t16 = (_Float16*)w;          w += (size_t)Bn * Sn * Dn * 2;      // 64 MB
    _Float16* s16 = (_Float16*)w;          w += (size_t)Bn * Sn * Dn * 2;      // 64 MB
    float* Cmat   = (float*)w;             w += (size_t)Bn * Sn * Sn * 4;      // 64 MB
    float* rnT    = (float*)w;             w += (size_t)Bn * Sn * 4;
    float* rnS    = (float*)w;             w += (size_t)Bn * Sn * 4;
    float* fg     = (float*)w;             w += (size_t)2 * Bn * Sn * 4;
    float* fv     = fg;
    float* gv     = fg + Bn * Sn;
    float* gpm    = (float*)w;             w += (size_t)Bn * NCHUNK * Sn * 4;  // 1 MB
    float* gps    = (float*)w;             w += (size_t)Bn * NCHUNK * Sn * 4;  // 1 MB
    float* rowpart= (float*)w;             w += (size_t)Bn * Sn * 4;

    const int rowBlocks = Bn * Sn / 8;     // 2048 (1 wave per row)

    // 1) normalize + f16 convert
    norm_convert<<<rowBlocks, 256, 0, stream>>>(teacher, t16, rnT);
    norm_convert<<<rowBlocks, 256, 0, stream>>>(student, s16, rnS);

    // 2) batched WMMA cost GEMM
    cost_gemm<<<dim3(Sn / 128, Sn / 64, Bn), 256, 0, stream>>>(t16, s16, rnT, rnS, Cmat);

    // 3) f = g = 0
    init_fg<<<(2 * Bn * Sn + 255) / 256, 256, 0, stream>>>(fg);

    // 4) 100 log-domain Sinkhorn iterations
    for (int it = 0; it < 100; ++it) {
        f_update<<<rowBlocks, 256, 0, stream>>>(Cmat, gv, fv);
        g_partial<<<Bn * NCHUNK, 256, 0, stream>>>(Cmat, fv, gpm, gps);
        g_combine<<<Bn, 256, 0, stream>>>(gpm, gps, gv);
    }

    // 5) loss = mean_b sum_ij P*C
    loss_rows<<<rowBlocks, 256, 0, stream>>>(Cmat, fv, gv, rowpart);
    loss_final<<<1, 256, 0, stream>>>(rowpart, out);
}
